// PPO_Actor_75144747810996
// MI455X (gfx1250) — compile-verified, matrix operand order checked by reading, not run on hardware
//
#include <hip/hip_runtime.h>
#include <hip/hip_bf16.h>

// ---------------------------------------------------------------------------
// Problem constants
// ---------------------------------------------------------------------------
#define BB 64
#define TT 128
#define NN 8
#define OO 128
#define FF 512
#define HH 256
#define AA 32
#define CC 64
#define VV 1024
#define BN (BB * NN)          // 512
#define MM (BN * TT)          // 65536 tokens

typedef _Float16 v8h  __attribute__((ext_vector_type(8)));
typedef _Float16 v16h __attribute__((ext_vector_type(16)));
typedef float    v8f  __attribute__((ext_vector_type(8)));

// ---------------------------------------------------------------------------
// Generic WMMA GEMM:  C[M,N] = act( A[M,K] @ W[N,K]^T + bias[N] (+ Src) )
// A, W row-major f16.  One wave computes a 16 x (16*NT) strip.
// blockDim = 128 (4 waves), grid = (N/(16*NT), M/64).
// ---------------------------------------------------------------------------
template<int NT, int ACT, bool W16, bool W32, bool ADDSRC>
__global__ __launch_bounds__(128)
void gemm_wmma(const _Float16* __restrict__ A, const _Float16* __restrict__ W,
               const float* __restrict__ bias,
               const _Float16* __restrict__ src, long srcStride,
               float* __restrict__ C32, _Float16* __restrict__ C16,
               int M, int N, int K)
{
    const int wave  = threadIdx.x >> 5;
    const int lane  = threadIdx.x & 31;
    const int half  = lane >> 4;       // which K-half of the fragment this lane holds
    const int lsub  = lane & 15;       // row (A) / col (B,D) within the 16-tile
    const int mtile = blockIdx.y * 4 + wave;
    const int ntb   = blockIdx.x * NT;
    if (mtile * 16 >= M) return;       // uniform per wave -> EXEC stays all-ones

    const size_t am = (size_t)(mtile * 16 + lsub);

    v8f acc[NT];
#pragma unroll
    for (int t = 0; t < NT; ++t) {
        float bv = bias ? bias[(ntb + t) * 16 + lsub] : 0.0f;
#pragma unroll
        for (int v = 0; v < 8; ++v) acc[t][v] = bv;
    }

    const int nkb = K >> 5;            // K blocks of 32
    for (int kb = 0; kb < nkb; ++kb) {
        const int k0 = (kb << 5) + (half << 3);
        // A fragment (16x32 f16): lane holds row am, K = k0..k0+7 and k0+16..k0+23
        v8h alo = *(const v8h*)(A + am * (size_t)K + k0);
        v8h ahi = *(const v8h*)(A + am * (size_t)K + k0 + 16);
        v16h af;
#pragma unroll
        for (int i = 0; i < 8; ++i) { af[i] = alo[i]; af[i + 8] = ahi[i]; }

#pragma unroll
        for (int t = 0; t < NT; ++t) {
            const size_t wn = (size_t)((ntb + t) * 16 + lsub);
            v8h blo = *(const v8h*)(W + wn * (size_t)K + k0);
            v8h bhi = *(const v8h*)(W + wn * (size_t)K + k0 + 16);
            v16h bf;
#pragma unroll
            for (int i = 0; i < 8; ++i) { bf[i] = blo[i]; bf[i + 8] = bhi[i]; }
            acc[t] = __builtin_amdgcn_wmma_f32_16x16x32_f16(
                false, af, false, bf, (short)0, acc[t], false, false);
        }
    }

    // D layout: lane holds col = lsub, rows m = v + 8*half  (v = 0..7)
#pragma unroll
    for (int t = 0; t < NT; ++t) {
        const int col = (ntb + t) * 16 + lsub;
#pragma unroll
        for (int v = 0; v < 8; ++v) {
            const size_t row = (size_t)(mtile * 16 + v + half * 8);
            float val = acc[t][v];
            if (ADDSRC) val += (float)src[row * (size_t)srcStride + col];
            if (ACT == 1) val = val > 0.0f ? val : 0.0f;
            if (W32) C32[row * (size_t)N + col] = val;
            if (W16) C16[row * (size_t)N + col] = (_Float16)val;
        }
    }
}

// ---------------------------------------------------------------------------
// Pointwise / helper kernels
// ---------------------------------------------------------------------------
__global__ void f32_to_f16(const float* __restrict__ s, _Float16* __restrict__ d, long n)
{
    long i = (long)blockIdx.x * blockDim.x + threadIdx.x;
    if (i < n) d[i] = (_Float16)s[i];
}

// x [B,T,N,O] fp32 -> X0 [(B*N)*T, O] f16
__global__ void transpose_x(const float* __restrict__ x, _Float16* __restrict__ X0)
{
    long i = (long)blockIdx.x * blockDim.x + threadIdx.x;
    if (i >= (long)BB * TT * NN * OO) return;
    int o = (int)(i & 127);
    int n = (int)((i >> 7) & 7);
    int t = (int)((i >> 10) & 127);
    int b = (int)(i >> 17);
    long dst = (((long)(b * NN + n) * TT) + t) * OO + o;
    X0[dst] = (_Float16)x[i];
}

__global__ void prep_small(const float* __restrict__ b_ih, const float* __restrict__ b_hh,
                           const float* __restrict__ b_a,  const float* __restrict__ b_c,
                           float* __restrict__ bsum, float* __restrict__ bcat,
                           float* __restrict__ lossacc)
{
    int i = threadIdx.x;
    if (i < 4 * HH) bsum[i] = b_ih[i] + b_hh[i];
    if (i < AA + CC) bcat[i] = (i < AA) ? b_a[i] : b_c[i - AA];
    if (i == 0) *lossacc = 0.0f;
}

__global__ void init_hc(const float* __restrict__ h0, const float* __restrict__ c0,
                        _Float16* __restrict__ h16, float* __restrict__ c32)
{
    int i = blockIdx.x * blockDim.x + threadIdx.x;
    if (i < BN * HH) { h16[i] = (_Float16)h0[i]; c32[i] = c0[i]; }
}

__device__ __forceinline__ float sigf(float x) { return 1.0f / (1.0f + expf(-x)); }

// gates [512,1024] fp32 (i,f,g,o) -> update c (f32), h (f16), Hseq, h_out/c_out
__global__ void lstm_pointwise(const float* __restrict__ gates,
                               float* __restrict__ c32, _Float16* __restrict__ h16,
                               _Float16* __restrict__ Hseq, int t,
                               float* __restrict__ hout, float* __restrict__ cout)
{
    int i = blockIdx.x * blockDim.x + threadIdx.x;
    if (i >= BN * HH) return;
    int r = i >> 8, j = i & (HH - 1);
    size_t g = (size_t)r * (4 * HH) + j;
    float ig = gates[g];
    float fg = gates[g + HH];
    float gg = gates[g + 2 * HH];
    float og = gates[g + 3 * HH];
    float c  = sigf(fg) * c32[i] + sigf(ig) * tanhf(gg);
    float h  = sigf(og) * tanhf(c);
    c32[i] = c;
    h16[i] = (_Float16)h;
    Hseq[((size_t)r * TT + t) * HH + j] = (_Float16)h;
    hout[i] = h;     // last step's write survives
    cout[i] = c;
}

// HeadOut [MM,96] fp32 -> action logits (d_out order b,t,n) + tanh comm flat
__global__ void head_pointwise(const float* __restrict__ ho,
                               float* __restrict__ act_out, float* __restrict__ comm_flat)
{
    long i = (long)blockIdx.x * blockDim.x + threadIdx.x;
    if (i >= (long)MM * (AA + CC)) return;
    int col = (int)(i % (AA + CC));
    long rseq = i / (AA + CC);             // bn*T + t
    int bn = (int)(rseq >> 7), t = (int)(rseq & 127);
    int b = bn >> 3, n = bn & 7;
    long rout = ((long)(b * TT + t)) * NN + n;
    float val = ho[i];
    if (col < AA) act_out[rout * AA + col] = val;
    else          comm_flat[rout * CC + (col - AA)] = tanhf(val);
}

// VQ: one wave per row; each lane scans 32 codes fully; butterfly argmin.
__global__ __launch_bounds__(256)
void vq_kernel(const float* __restrict__ flat, const float* __restrict__ cb,
               float* __restrict__ qout, float* __restrict__ idxout,
               float* __restrict__ lossacc)
{
    __shared__ float srow[8][CC];
    int wave = threadIdx.x >> 5, lane = threadIdx.x & 31;
    int r = blockIdx.x * 8 + wave;
    srow[wave][lane]      = flat[(size_t)r * CC + lane];
    srow[wave][lane + 32] = flat[(size_t)r * CC + 32 + lane];

    float minv = 3.4e38f; int mini = 0;
    for (int vi = 0; vi < VV / 32; ++vi) {
        int v = vi * 32 + lane;
        const float* c = cb + (size_t)v * CC;
        float d = 0.0f;
#pragma unroll 8
        for (int k = 0; k < CC; ++k) {
            float df = srow[wave][k] - c[k];
            d = fmaf(df, df, d);
        }
        if (d < minv || (d == minv && v < mini)) { minv = d; mini = v; }
    }
    for (int off = 16; off > 0; off >>= 1) {       // butterfly: all lanes get argmin
        float ov = __shfl_xor(minv, off);
        int   oi = __shfl_xor(mini, off);
        if (ov < minv || (ov == minv && oi < mini)) { minv = ov; mini = oi; }
    }
    if (lane == 0) idxout[r] = (float)mini;

    float s = 0.0f;
    for (int k = lane; k < CC; k += 32) {
        float qv = cb[(size_t)mini * CC + k];
        qout[(size_t)r * CC + k] = qv;             // straight-through: out == q
        float df = qv - srow[wave][k];
        s += df * df;
    }
    for (int off = 16; off > 0; off >>= 1) s += __shfl_xor(s, off);
    if (lane == 0) atomicAdd(lossacc, s);
}

__global__ void finalize(const float* __restrict__ acc,
                         float* __restrict__ cbl, float* __restrict__ loss)
{
    float l = *acc / (float)((long)MM * CC);
    *cbl  = l;                 // codebook_loss == e_latent_loss numerically
    *loss = 1.25f * l;         // + COMMITMENT_COST * e_latent
}

// ---------------------------------------------------------------------------
// Host launcher
// ---------------------------------------------------------------------------
extern "C" void kernel_launch(void* const* d_in, const int* in_sizes, int n_in,
                              void* d_out, int out_size, void* d_ws, size_t ws_size,
                              hipStream_t stream)
{
    const float* x    = (const float*)d_in[0];
    const float* h0   = (const float*)d_in[1];
    const float* c0   = (const float*)d_in[2];
    const float* w1   = (const float*)d_in[3];
    const float* b1   = (const float*)d_in[4];
    const float* w2   = (const float*)d_in[5];
    const float* b2   = (const float*)d_in[6];
    const float* w3   = (const float*)d_in[7];
    const float* b3   = (const float*)d_in[8];
    const float* w_ih = (const float*)d_in[9];
    const float* w_hh = (const float*)d_in[10];
    const float* b_ih = (const float*)d_in[11];
    const float* b_hh = (const float*)d_in[12];
    const float* w_a  = (const float*)d_in[13];
    const float* b_a  = (const float*)d_in[14];
    const float* w_c  = (const float*)d_in[15];
    const float* b_c  = (const float*)d_in[16];
    const float* cb   = (const float*)d_in[17];

    // d_out offsets (floats), in reference return order
    float* out = (float*)d_out;
    const long OFF_ACT  = 0;
    const long OFF_COMM = OFF_ACT  + (long)BB * TT * NN * AA;   // 2097152
    const long OFF_CBL  = OFF_COMM + (long)BB * TT * NN * CC;   // 6291456
    const long OFF_IDX  = OFF_CBL  + 1;
    const long OFF_LOSS = OFF_IDX  + (long)MM;
    const long OFF_H    = OFF_LOSS + 1;
    const long OFF_C    = OFF_H    + (long)BN * HH;

    // workspace carving (256B aligned)
    char* p = (char*)d_ws;
    auto carve = [&](size_t bytes) -> void* {
        void* r = (void*)p;
        p += (bytes + 255) & ~(size_t)255;
        return r;
    };
    _Float16* X0    = (_Float16*)carve((size_t)MM * OO * 2);
    _Float16* Ya    = (_Float16*)carve((size_t)MM * FF * 2);
    _Float16* Yb    = (_Float16*)carve((size_t)MM * FF * 2);
    _Float16* G16   = (_Float16*)carve((size_t)MM * 4 * HH * 2);
    float*    gates = (float*)   carve((size_t)BN * 4 * HH * 4);
    _Float16* h16   = (_Float16*)carve((size_t)BN * HH * 2);
    float*    c32   = (float*)   carve((size_t)BN * HH * 4);
    _Float16* Hseq  = (_Float16*)carve((size_t)MM * HH * 2);
    _Float16* w1f   = (_Float16*)carve((size_t)FF * OO * 2);
    _Float16* w2f   = (_Float16*)carve((size_t)FF * FF * 2);
    _Float16* w3f   = (_Float16*)carve((size_t)FF * FF * 2);
    _Float16* wihf  = (_Float16*)carve((size_t)4 * HH * FF * 2);
    _Float16* whhf  = (_Float16*)carve((size_t)4 * HH * HH * 2);
    _Float16* Wcat  = (_Float16*)carve((size_t)(AA + CC) * HH * 2);
    float*    bsum  = (float*)   carve((size_t)4 * HH * 4);
    float*    bcat  = (float*)   carve((size_t)(AA + CC) * 4);
    float*    HeadO = (float*)   carve((size_t)MM * (AA + CC) * 4);
    float*    cflat = (float*)   carve((size_t)MM * CC * 4);
    float*    lacc  = (float*)   carve(256);

    auto cvt = [&](const float* s, _Float16* d, long n) {
        f32_to_f16<<<dim3((unsigned)((n + 255) / 256)), 256, 0, stream>>>(s, d, n);
    };

    // --- prep: biases, f16 weights, state init, input transpose -------------
    prep_small<<<1, 1024, 0, stream>>>(b_ih, b_hh, b_a, b_c, bsum, bcat, lacc);
    cvt(w1,   w1f,  (long)FF * OO);
    cvt(w2,   w2f,  (long)FF * FF);
    cvt(w3,   w3f,  (long)FF * FF);
    cvt(w_ih, wihf, (long)4 * HH * FF);
    cvt(w_hh, whhf, (long)4 * HH * HH);
    cvt(w_a,  Wcat,               (long)AA * HH);
    cvt(w_c,  Wcat + AA * HH,     (long)CC * HH);
    init_hc<<<dim3((BN * HH + 255) / 256), 256, 0, stream>>>(h0, c0, h16, c32);
    transpose_x<<<dim3((unsigned)(((long)MM * OO + 255) / 256)), 256, 0, stream>>>(x, X0);

    // --- MLP body: three relu GEMMs -----------------------------------------
    gemm_wmma<4, 1, true, false, false><<<dim3(FF / 64, MM / 64), 128, 0, stream>>>(
        X0, w1f, b1, nullptr, 0, nullptr, Ya, MM, FF, OO);
    gemm_wmma<4, 1, true, false, false><<<dim3(FF / 64, MM / 64), 128, 0, stream>>>(
        Ya, w2f, b2, nullptr, 0, nullptr, Yb, MM, FF, FF);
    gemm_wmma<4, 1, true, false, false><<<dim3(FF / 64, MM / 64), 128, 0, stream>>>(
        Yb, w3f, b3, nullptr, 0, nullptr, Ya, MM, FF, FF);

    // --- LSTM input projection hoisted over all T: G = Y3 @ w_ih^T + bsum ---
    gemm_wmma<4, 0, true, false, false><<<dim3(4 * HH / 64, MM / 64), 128, 0, stream>>>(
        Ya, wihf, bsum, nullptr, 0, nullptr, G16, MM, 4 * HH, FF);

    // --- recurrence: 128 sequential steps ------------------------------------
    for (int t = 0; t < TT; ++t) {
        gemm_wmma<4, 0, false, true, true><<<dim3(4 * HH / 64, BN / 64), 128, 0, stream>>>(
            h16, whhf, nullptr, G16 + (size_t)t * 4 * HH, (long)TT * 4 * HH,
            gates, nullptr, BN, 4 * HH, HH);
        lstm_pointwise<<<dim3((BN * HH + 255) / 256), 256, 0, stream>>>(
            gates, c32, h16, Hseq, t, out + OFF_H, out + OFF_C);
    }

    // --- heads: concatenated [96,256] GEMM -----------------------------------
    gemm_wmma<2, 0, false, true, false><<<dim3((AA + CC) / 32, MM / 64), 128, 0, stream>>>(
        Hseq, Wcat, bcat, nullptr, 0, HeadO, nullptr, MM, AA + CC, HH);
    head_pointwise<<<dim3((unsigned)(((long)MM * (AA + CC) + 255) / 256)), 256, 0, stream>>>(
        HeadO, out + OFF_ACT, cflat);

    // --- VQ + losses ----------------------------------------------------------
    vq_kernel<<<dim3(MM / 8), 256, 0, stream>>>(
        cflat, cb, out + OFF_COMM, out + OFF_IDX, lacc);
    finalize<<<1, 1, 0, stream>>>(lacc, out + OFF_CBL, out + OFF_LOSS);

    (void)in_sizes; (void)n_in; (void)out_size; (void)ws_size;
}
